// WhaleAudioEncoderLayer_17291538333978
// MI455X (gfx1250) — matrix operand order repaired and checked by reading, not compiled, not run on hardware
//
#include <hip/hip_runtime.h>
#include <cstdint>
#include <cstddef>

// ---------------- problem constants (from reference) ----------------
#define N_SEQ  2048
#define C_DIM  1024
#define H_HEADS 16
#define D_HEAD  64
#define FF_DIM 4096
#define LN_EPS 1e-5f
#define ATT_SCALE 0.125f   // 64^-0.5

typedef __attribute__((ext_vector_type(16))) _Float16 v16h;
typedef __attribute__((ext_vector_type(8)))  float    v8f;

union Frag16 { v16h h; uint4 u[2]; };

// =====================================================================
// Register-blocked WMMA GEMM:
//   D[M x Nout] = A[M x K](f16) * B[Nout x K](f16)^T
//   + optional bias[Nout], + optional fp32 residual (ldr),
//   optional relu, optional fp32 accumulate into D, f16 or f32 output.
// One wave computes a (AM*16) x (BN*16) tile: AM*BN wmma per K-step of 32,
// reusing each A fragment BN times and each B fragment AM times.
// 8 waves / 256-thread block; tile id is wave-uniform so EXEC stays all-1s
// for every executed WMMA. K % 32 == 0, M % (AM*16) == 0, Nout % (BN*16)==0.
// =====================================================================
template<int AM, int BN>
__global__ void __launch_bounds__(256)
gemm_wmma_f16(const _Float16* __restrict__ A, int lda,
              const _Float16* __restrict__ B, int ldb,
              void* __restrict__ Dp, int ldd, int out_f16,
              int M, int Nout, int K,
              const float* __restrict__ bias,
              const float* __restrict__ resid, int ldr,
              int accum, int relu)
{
    const int lane = threadIdx.x & 31;
    const int wave = threadIdx.x >> 5;
    const int tilesX = Nout / (BN * 16);
    const int total  = tilesX * (M / (AM * 16));
    const int gw = blockIdx.x * 8 + wave;        // wave-uniform
    if (gw >= total) return;                     // whole wave exits together
    const int tn = (gw % tilesX) * (BN * 16);
    const int tm = (gw / tilesX) * (AM * 16);

    const int half = lane >> 4;                  // ISA lane-half (K split)
    const int l15  = lane & 15;                  // M (for A) / N (for B) index

    // A frag (ISA 7.12.2, 16-bit A 16x32): lane l15 = row; half 0 -> K {0..7,16..23},
    // half 1 -> K {8..15,24..31}. Pre-offset by half*8 so chunks are +k0 and +k0+16.
    const _Float16* arow[AM];
    #pragma unroll
    for (int i = 0; i < AM; ++i)
        arow[i] = A + (size_t)(tm + i * 16 + l15) * lda + half * 8;
    // B frag: lane l15 = output column (row of [Nout x K] weight); half 0 -> K {0..7,16..23},
    // half 1 -> K {8..15,24..31} (same as A). Pre-offset by half*8.
    const _Float16* brow[BN];
    #pragma unroll
    for (int j = 0; j < BN; ++j)
        brow[j] = B + (size_t)(tn + j * 16 + l15) * ldb + half * 8;

    v8f acc[AM][BN];
    #pragma unroll
    for (int i = 0; i < AM; ++i)
        #pragma unroll
        for (int j = 0; j < BN; ++j)
            acc[i][j] = (v8f){};

    for (int k0 = 0; k0 < K; k0 += 32) {
        if (k0 + 64 < K) {                       // stream next lines into GL2/L0
            #pragma unroll
            for (int i = 0; i < AM; ++i) __builtin_prefetch(arow[i] + k0 + 64, 0, 1);
            #pragma unroll
            for (int j = 0; j < BN; ++j) __builtin_prefetch(brow[j] + k0 + 64, 0, 1);
        }
        Frag16 af[AM], bf[BN];
        #pragma unroll
        for (int i = 0; i < AM; ++i) {
            af[i].u[0] = *(const uint4*)(arow[i] + k0);
            af[i].u[1] = *(const uint4*)(arow[i] + k0 + 16);
        }
        #pragma unroll
        for (int j = 0; j < BN; ++j) {
            bf[j].u[0] = *(const uint4*)(brow[j] + k0);
            bf[j].u[1] = *(const uint4*)(brow[j] + k0 + 16);
        }
        #pragma unroll
        for (int i = 0; i < AM; ++i)
            #pragma unroll
            for (int j = 0; j < BN; ++j)
                acc[i][j] = __builtin_amdgcn_wmma_f32_16x16x32_f16(
                    false, af[i].h, false, bf[j].h, (short)0, acc[i][j], false, false);
    }

    // D layout: VGPR r -> row + r + 8*half, column l15
    #pragma unroll
    for (int j = 0; j < BN; ++j) {
        const int col = tn + j * 16 + l15;
        const float badd = bias ? bias[col] : 0.0f;
        #pragma unroll
        for (int i = 0; i < AM; ++i) {
            #pragma unroll
            for (int r = 0; r < 8; ++r) {
                const int row = tm + i * 16 + r + half * 8;
                const size_t idx = (size_t)row * ldd + col;
                float v = acc[i][j][r] + badd;
                if (resid) v += resid[(size_t)row * ldr + col];
                if (out_f16) {
                    if (relu) v = fmaxf(v, 0.0f);
                    ((_Float16*)Dp)[idx] = (_Float16)v;
                } else {
                    float* Df = (float*)Dp;
                    if (accum) v += Df[idx];
                    if (relu) v = fmaxf(v, 0.0f);
                    Df[idx] = v;
                }
            }
        }
    }
}

// ---------------- LayerNorm over C, fp32 in -> f16 out ----------------
__global__ void __launch_bounds__(256)
layernorm_f16(const float* __restrict__ X, const float* __restrict__ g,
              const float* __restrict__ be, _Float16* __restrict__ out)
{
    __shared__ float s1[256], s2[256];
    const int row = blockIdx.x;
    const float* x = X + (size_t)row * C_DIM;
    float a = 0.f, b = 0.f;
    for (int c = threadIdx.x; c < C_DIM; c += 256) { float v = x[c]; a += v; b += v * v; }
    s1[threadIdx.x] = a; s2[threadIdx.x] = b; __syncthreads();
    for (int st = 128; st > 0; st >>= 1) {
        if (threadIdx.x < st) { s1[threadIdx.x] += s1[threadIdx.x + st];
                                s2[threadIdx.x] += s2[threadIdx.x + st]; }
        __syncthreads();
    }
    const float mean = s1[0] * (1.0f / C_DIM);
    const float var  = s2[0] * (1.0f / C_DIM) - mean * mean;
    const float rstd = rsqrtf(var + LN_EPS);
    for (int c = threadIdx.x; c < C_DIM; c += 256)
        out[(size_t)row * C_DIM + c] = (_Float16)((x[c] - mean) * rstd * g[c] + be[c]);
}

// ---------------- masked, scaled softmax over one row of N_SEQ --------
__global__ void __launch_bounds__(256)
softmax_row_f16(const float* __restrict__ S, const unsigned char* __restrict__ mask,
                _Float16* __restrict__ out)
{
    __shared__ float red[256];
    const int row = blockIdx.x;
    const int t = threadIdx.x;
    const float* s = S + (size_t)row * N_SEQ;
    _Float16* o = out + (size_t)row * N_SEQ;
    const float NEG = -3.0e38f;

    float l[N_SEQ / 256];
    float mx = NEG;
    #pragma unroll
    for (int i = 0; i < N_SEQ / 256; ++i) {
        const int c = t + i * 256;
        const float v = mask[c] ? s[c] * ATT_SCALE : NEG;
        l[i] = v; mx = fmaxf(mx, v);
    }
    red[t] = mx; __syncthreads();
    for (int st = 128; st > 0; st >>= 1) {
        if (t < st) red[t] = fmaxf(red[t], red[t + st]);
        __syncthreads();
    }
    mx = red[0]; __syncthreads();

    float sum = 0.f;
    #pragma unroll
    for (int i = 0; i < N_SEQ / 256; ++i) { float e = __expf(l[i] - mx); l[i] = e; sum += e; }
    red[t] = sum; __syncthreads();
    for (int st = 128; st > 0; st >>= 1) {
        if (t < st) red[t] += red[t + st];
        __syncthreads();
    }
    const float inv = 1.0f / red[0];
    #pragma unroll
    for (int i = 0; i < N_SEQ / 256; ++i) {
        const int c = t + i * 256;
        o[c] = (_Float16)(l[i] * inv);
    }
}

// ---------------- fp32 -> f16 elementwise ----------------
__global__ void cvt_f32_f16_k(const float* __restrict__ in, _Float16* __restrict__ out, size_t n)
{
    size_t i = (size_t)blockIdx.x * blockDim.x + threadIdx.x;
    if (i < n) out[i] = (_Float16)in[i];
}

// ---------------- bias combine: out = a + b ----------------
__global__ void add_vec_k(const float* __restrict__ a, const float* __restrict__ b,
                          float* __restrict__ out, int n)
{
    int i = blockIdx.x * blockDim.x + threadIdx.x;
    if (i < n) out[i] = a[i] + b[i];
}

// ---------------- V transpose: v16[N][C] -> vT16[C][N] (== [H][Dh][N]) ---
__global__ void transpose_v_k(const _Float16* __restrict__ v, _Float16* __restrict__ vt)
{
    size_t o = (size_t)blockIdx.x * blockDim.x + threadIdx.x;
    if (o >= (size_t)N_SEQ * C_DIM) return;
    const size_t dall = o / N_SEQ;     // h*Dh + d
    const size_t m    = o % N_SEQ;
    vt[o] = v[m * C_DIM + dall];
}

// =====================================================================
extern "C" void kernel_launch(void* const* d_in, const int* in_sizes, int n_in,
                              void* d_out, int out_size, void* d_ws, size_t ws_size,
                              hipStream_t stream)
{
    (void)in_sizes; (void)n_in; (void)out_size; (void)ws_size;
    const float* hidden = (const float*)d_in[0];
    const unsigned char* mask = (const unsigned char*)d_in[1];
    const float* pos  = (const float*)d_in[2];
    const float* Wq   = (const float*)d_in[3];
    const float* bq   = (const float*)d_in[4];
    const float* Wk   = (const float*)d_in[5];
    const float* bk   = (const float*)d_in[6];
    const float* Wv   = (const float*)d_in[7];
    const float* bv   = (const float*)d_in[8];
    const float* Wo   = (const float*)d_in[9];
    const float* bo   = (const float*)d_in[10];
    const float* Wp   = (const float*)d_in[11];
    const float* pbu  = (const float*)d_in[12];
    const float* pbv  = (const float*)d_in[13];
    const float* W1   = (const float*)d_in[14];
    const float* b1   = (const float*)d_in[15];
    const float* W2   = (const float*)d_in[16];
    const float* b2   = (const float*)d_in[17];
    const float* g1   = (const float*)d_in[18];
    const float* be1  = (const float*)d_in[19];
    const float* g2   = (const float*)d_in[20];
    const float* be2  = (const float*)d_in[21];
    float* out = (float*)d_out;

    // ---- workspace carve-up (256B aligned) ----
    char* ws = (char*)d_ws;
    size_t off = 0;
    auto take = [&](size_t bytes) -> char* {
        char* p = ws + off;
        off = (off + bytes + 255) & ~(size_t)255;
        return p;
    };
    const size_t NC = (size_t)N_SEQ * C_DIM;
    _Float16* xn16   = (_Float16*)take(NC * 2);
    _Float16* Wq16   = (_Float16*)take((size_t)C_DIM * C_DIM * 2);
    _Float16* Wk16   = (_Float16*)take((size_t)C_DIM * C_DIM * 2);
    _Float16* Wv16   = (_Float16*)take((size_t)C_DIM * C_DIM * 2);
    _Float16* Wo16   = (_Float16*)take((size_t)C_DIM * C_DIM * 2);
    _Float16* Wp16   = (_Float16*)take((size_t)C_DIM * C_DIM * 2);
    _Float16* W1_16  = (_Float16*)take((size_t)FF_DIM * C_DIM * 2);
    _Float16* W2_16  = (_Float16*)take((size_t)C_DIM * FF_DIM * 2);
    _Float16* pos16  = (_Float16*)take(NC * 2);
    float*    bu_t   = (float*)take(C_DIM * 4);
    float*    bv_t   = (float*)take(C_DIM * 4);
    _Float16* qu16   = (_Float16*)take(NC * 2);
    _Float16* qv16   = (_Float16*)take(NC * 2);
    _Float16* k16    = (_Float16*)take(NC * 2);
    _Float16* p16    = (_Float16*)take(NC * 2);
    _Float16* v16    = (_Float16*)take(NC * 2);
    _Float16* vT16   = (_Float16*)take(NC * 2);
    float*    sc32   = (float*)take((size_t)N_SEQ * N_SEQ * 4);
    _Float16* attn16 = (_Float16*)take((size_t)N_SEQ * N_SEQ * 2);
    _Float16* ao16   = (_Float16*)take(NC * 2);
    float*    h2     = (float*)take(NC * 4);
    _Float16* xn2_16 = (_Float16*)take(NC * 2);
    _Float16* ff1_16 = (_Float16*)take((size_t)N_SEQ * FF_DIM * 2);

    auto cvt = [&](const float* i, _Float16* o, size_t n) {
        cvt_f32_f16_k<<<(unsigned)((n + 255) / 256), 256, 0, stream>>>(i, o, n);
    };
    // big-tile GEMM: 64x64 per wave (16 wmma / 8 frag loads per K-step)
    auto gemm = [&](const _Float16* A, int lda, const _Float16* B, int ldb,
                    void* D, int ldd, int d16, int M, int Nout, int K,
                    const float* bias, const float* resid, int ldr,
                    int accum, int relu) {
        const int total = (M / 64) * (Nout / 64);
        gemm_wmma_f16<4, 4><<<(total + 7) / 8, 256, 0, stream>>>(
            A, lda, B, ldb, D, ldd, d16, M, Nout, K, bias, resid, ldr, accum, relu);
    };
    // small-N GEMM (attn . V, Nout = 64): 16x16 tiles keep 512 waves in flight
    auto gemm_s = [&](const _Float16* A, int lda, const _Float16* B, int ldb,
                      void* D, int ldd, int d16, int M, int Nout, int K) {
        const int total = (M / 16) * (Nout / 16);
        gemm_wmma_f16<1, 1><<<(total + 7) / 8, 256, 0, stream>>>(
            A, lda, B, ldb, D, ldd, d16, M, Nout, K, nullptr, nullptr, 0, 0, 0);
    };

    // ---- weight / input conversions ----
    cvt(Wq, Wq16, (size_t)C_DIM * C_DIM);
    cvt(Wk, Wk16, (size_t)C_DIM * C_DIM);
    cvt(Wv, Wv16, (size_t)C_DIM * C_DIM);
    cvt(Wo, Wo16, (size_t)C_DIM * C_DIM);
    cvt(Wp, Wp16, (size_t)C_DIM * C_DIM);
    cvt(W1, W1_16, (size_t)FF_DIM * C_DIM);
    cvt(W2, W2_16, (size_t)C_DIM * FF_DIM);
    cvt(pos, pos16, NC);
    add_vec_k<<<C_DIM / 256, 256, 0, stream>>>(bq, pbu, bu_t, C_DIM);
    add_vec_k<<<C_DIM / 256, 256, 0, stream>>>(bq, pbv, bv_t, C_DIM);

    // ---- pre-norm ----
    layernorm_f16<<<N_SEQ, 256, 0, stream>>>(hidden, g1, be1, xn16);

    // ---- projections (pos_bias folded into Q bias) ----
    gemm(xn16, C_DIM, Wq16, C_DIM, qu16, C_DIM, 1, N_SEQ, C_DIM, C_DIM, bu_t, nullptr, 0, 0, 0);
    gemm(xn16, C_DIM, Wq16, C_DIM, qv16, C_DIM, 1, N_SEQ, C_DIM, C_DIM, bv_t, nullptr, 0, 0, 0);
    gemm(xn16, C_DIM, Wk16, C_DIM, k16,  C_DIM, 1, N_SEQ, C_DIM, C_DIM, bk,   nullptr, 0, 0, 0);
    gemm(xn16, C_DIM, Wv16, C_DIM, v16,  C_DIM, 1, N_SEQ, C_DIM, C_DIM, bv,   nullptr, 0, 0, 0);
    gemm(pos16, C_DIM, Wp16, C_DIM, p16, C_DIM, 1, N_SEQ, C_DIM, C_DIM, nullptr, nullptr, 0, 0, 0);
    transpose_v_k<<<(unsigned)((NC + 255) / 256), 256, 0, stream>>>(v16, vT16);

    // ---- attention, per head (score buffer reused) ----
    for (int h = 0; h < H_HEADS; ++h) {
        const _Float16* quh = qu16 + h * D_HEAD;
        const _Float16* qvh = qv16 + h * D_HEAD;
        const _Float16* kh  = k16  + h * D_HEAD;
        const _Float16* ph  = p16  + h * D_HEAD;
        const _Float16* vth = vT16 + (size_t)h * D_HEAD * N_SEQ;
        // matrix_ac = qu . k^T
        gemm(quh, C_DIM, kh, C_DIM, sc32, N_SEQ, 0, N_SEQ, N_SEQ, D_HEAD,
             nullptr, nullptr, 0, 0, 0);
        // + matrix_bd = qv . p^T  (fp32 accumulate)
        gemm(qvh, C_DIM, ph, C_DIM, sc32, N_SEQ, 0, N_SEQ, N_SEQ, D_HEAD,
             nullptr, nullptr, 0, 1, 0);
        // scale + mask + softmax -> f16
        softmax_row_f16<<<N_SEQ, 256, 0, stream>>>(sc32, mask, attn16);
        // o_h = attn . V_h   (B = vT[h] rows are Dh, contiguous along K=m)
        gemm_s(attn16, N_SEQ, vth, N_SEQ, ao16 + h * D_HEAD, C_DIM, 1,
               N_SEQ, D_HEAD, N_SEQ);
    }

    // ---- output projection + residual ----
    gemm(ao16, C_DIM, Wo16, C_DIM, h2, C_DIM, 0, N_SEQ, C_DIM, C_DIM,
         bo, hidden, C_DIM, 0, 0);

    // ---- FFN with pre-norm + residual ----
    layernorm_f16<<<N_SEQ, 256, 0, stream>>>(h2, g2, be2, xn2_16);
    gemm(xn2_16, C_DIM, W1_16, C_DIM, ff1_16, FF_DIM, 1, N_SEQ, FF_DIM, C_DIM,
         b1, nullptr, 0, 0, 1);                               // relu, f16 out
    gemm(ff1_16, FF_DIM, W2_16, FF_DIM, out, C_DIM, 0, N_SEQ, C_DIM, FF_DIM,
         b2, h2, C_DIM, 0, 0);
}
